// LMHead_78821239816356
// MI455X (gfx1250) — matrix operand order repaired and checked by reading, not compiled, not run on hardware
//
#include <hip/hip_runtime.h>
#include <hip/hip_bf16.h>
#include <math.h>

// ---------------------------------------------------------------------------
// LM head for MI455X (gfx1250, wave32, WMMA):
//   logits = hidden[4096,1024] @ W[50257,1024]^T ; out = log_softmax(logits)
//
// Precision: fp32 inputs split into bf16 hi + lo (x = hi + lo). Product is
// hi*hi + hi*lo + lo*hi with fp32 WMMA accumulation -> near-fp32 accuracy.
// bf16 W-hi is 103 MB and fits the 192 MB L2, so W re-reads across the 32
// M-tiles are L2 hits. HBM floor ~2.5 GB (~107us @ 23.3 TB/s); bf16x3
// compute ~422 GFLOP x3 -> compute-bound at matrix-pipe rates, hence the
// double-buffered K pipeline below to keep the WMMA pipe saturated.
// ---------------------------------------------------------------------------

typedef __attribute__((ext_vector_type(16))) __bf16 v16bf;
typedef __attribute__((ext_vector_type(8)))  float  v8f;

#define VOCAB 50257
#define EMBED 1024
#define NROWS 4096   // 2 * 2048

union FragU {          // 32 bytes = 8 VGPRs = one 16-bit WMMA operand fragment
    v16bf v;
    uint4 q[2];
};

// ---------------------------------------------------------------------------
// Kernel 1: split fp32 -> bf16 hi + bf16 lo (grid-stride, bandwidth bound)
// ---------------------------------------------------------------------------
__global__ __launch_bounds__(256)
void split_bf16_kernel(const float* __restrict__ src,
                       __bf16* __restrict__ hi, __bf16* __restrict__ lo,
                       long n) {
    long i      = (long)blockIdx.x * blockDim.x + threadIdx.x;
    long stride = (long)gridDim.x * blockDim.x;
    for (; i < n; i += stride) {
        float  x = src[i];
        __bf16 h = (__bf16)x;
        float  r = x - (float)h;
        hi[i] = h;
        lo[i] = (__bf16)r;
    }
}

// ---------------------------------------------------------------------------
// Kernel 2: GEMM. Block = 256 threads = 8 waves arranged 2(M) x 4(N).
// Block tile 128x128, wave tile 64x32 = 4x2 WMMA 16x16 accumulators,
// K pipelined in steps of 64 with two named ping-pong fragment buffers.
//
// Fragment layouts (CDNA5 ISA 7.12.2, 16-bit operands, wave32), h = lane/16:
//   A (16x32, MxK): lane holds row M = lane%16; regs V0-3 = K 8h..8h+7,
//                   V4-7 = K 16+8h..16+8h+7  -> two b128 loads per frag.
//   B (32x16, KxN): lane holds col N = lane%16; regs V0-3 = K 8h..8h+7,
//                   V4-7 = K 16+8h..16+8h+7  -> two b128 loads per frag.
//   C/D (16x16)   : VGPR r, lanes 0-15 -> M=r, lanes 16-31 -> M=8+r; N=lane%16.
// ---------------------------------------------------------------------------

#define LOAD_FRAGS(buf_aHi, buf_aLo, buf_bHi, buf_bLo, kk)                     \
    do {                                                                       \
        _Pragma("unroll")                                                      \
        for (int i = 0; i < 4; ++i) {                                          \
            buf_aHi[i].q[0] = *(const uint4*)(aHiP[i] + (kk) + 8 * h);         \
            buf_aHi[i].q[1] = *(const uint4*)(aHiP[i] + (kk) + 16 + 8 * h);    \
            buf_aLo[i].q[0] = *(const uint4*)(aLoP[i] + (kk) + 8 * h);         \
            buf_aLo[i].q[1] = *(const uint4*)(aLoP[i] + (kk) + 16 + 8 * h);    \
        }                                                                      \
        _Pragma("unroll")                                                      \
        for (int j = 0; j < 2; ++j) {                                          \
            buf_bHi[j].q[0] = *(const uint4*)(bHiP[j] + (kk) + 8 * h);         \
            buf_bHi[j].q[1] = *(const uint4*)(bHiP[j] + (kk) + 16 + 8 * h);    \
            buf_bLo[j].q[0] = *(const uint4*)(bLoP[j] + (kk) + 8 * h);         \
            buf_bLo[j].q[1] = *(const uint4*)(bLoP[j] + (kk) + 16 + 8 * h);    \
        }                                                                      \
    } while (0)

#define MMA_FRAGS(buf_aHi, buf_aLo, buf_bHi, buf_bLo)                          \
    do {                                                                       \
        _Pragma("unroll")                                                      \
        for (int i = 0; i < 4; ++i) {                                          \
            _Pragma("unroll")                                                  \
            for (int j = 0; j < 2; ++j) {                                      \
                acc[i][j] = __builtin_amdgcn_wmma_f32_16x16x32_bf16(           \
                    false, buf_aHi[i].v, false, buf_bHi[j].v, (short)0,        \
                    acc[i][j], false, false);                                  \
                acc[i][j] = __builtin_amdgcn_wmma_f32_16x16x32_bf16(           \
                    false, buf_aHi[i].v, false, buf_bLo[j].v, (short)0,        \
                    acc[i][j], false, false);                                  \
                acc[i][j] = __builtin_amdgcn_wmma_f32_16x16x32_bf16(           \
                    false, buf_aLo[i].v, false, buf_bHi[j].v, (short)0,        \
                    acc[i][j], false, false);                                  \
            }                                                                  \
        }                                                                      \
    } while (0)

__global__ __launch_bounds__(256)
void lmhead_gemm_kernel(const __bf16* __restrict__ Whi,
                        const __bf16* __restrict__ Wlo,
                        const __bf16* __restrict__ Hhi,
                        const __bf16* __restrict__ Hlo,
                        float* __restrict__ out) {
    const int lane = threadIdx.x & 31;
    const int wave = threadIdx.x >> 5;   // 0..7
    const int wm   = wave >> 2;          // 0..1 : M direction
    const int wn   = wave & 3;           // 0..3 : N direction
    const int h    = lane >> 4;          // lane half
    const int l16  = lane & 15;

    const int mWave = blockIdx.y * 128 + wm * 64;   // NROWS divisible by 128
    const int nWave = blockIdx.x * 128 + wn * 32;

    // Per-lane base pointers for the 4 A row-subtiles and 2 B col-subtiles.
    const __bf16* aHiP[4];
    const __bf16* aLoP[4];
#pragma unroll
    for (int i = 0; i < 4; ++i) {
        const int row = mWave + 16 * i + l16;
        aHiP[i] = Hhi + (long)row * EMBED;
        aLoP[i] = Hlo + (long)row * EMBED;
    }
    const __bf16* bHiP[2];
    const __bf16* bLoP[2];
    int colN[2];
#pragma unroll
    for (int j = 0; j < 2; ++j) {
        int col  = nWave + 16 * j + l16;
        colN[j]  = col;                           // for store masking
        int colc = col < VOCAB ? col : VOCAB - 1; // clamp loads in-bounds
        bHiP[j] = Whi + (long)colc * EMBED;
        bLoP[j] = Wlo + (long)colc * EMBED;
    }

    v8f acc[4][2];
#pragma unroll
    for (int i = 0; i < 4; ++i)
#pragma unroll
        for (int j = 0; j < 2; ++j)
            acc[i][j] = {};

    // --- software-pipelined K loop: two named ping-pong buffers -------------
    FragU aHi0[4], aLo0[4], bHi0[2], bLo0[2];   // even 32-K steps
    FragU aHi1[4], aLo1[4], bHi1[2], bLo1[2];   // odd  32-K steps

    LOAD_FRAGS(aHi0, aLo0, bHi0, bLo0, 0);      // prologue: kt = 0

#pragma unroll 1
    for (int kt = 0; kt < EMBED; kt += 64) {
        // issue loads for the odd step, then do math on the even step
        LOAD_FRAGS(aHi1, aLo1, bHi1, bLo1, kt + 32);
        __builtin_prefetch(bHiP[0] + kt + 64 + 16 * h, 0, 1);
        __builtin_prefetch(bHiP[1] + kt + 64 + 16 * h, 0, 1);
        MMA_FRAGS(aHi0, aLo0, bHi0, bLo0);

        // issue loads for the next even step, then math on the odd step
        if (kt + 64 < EMBED) {
            LOAD_FRAGS(aHi0, aLo0, bHi0, bLo0, kt + 64);
        }
        MMA_FRAGS(aHi1, aLo1, bHi1, bLo1);
    }

    // Store: VGPR r -> row mWave+16i+(h?8:0)+r, col = nWave+16j+l16.
    // Lanes 0-15 of each register store 64 contiguous bytes (coalesced).
#pragma unroll
    for (int i = 0; i < 4; ++i) {
        const int rbase = mWave + 16 * i + 8 * h;
#pragma unroll
        for (int j = 0; j < 2; ++j) {
            if (colN[j] < VOCAB) {
                float* p = out + (long)rbase * VOCAB + colN[j];
#pragma unroll
                for (int r = 0; r < 8; ++r)
                    p[(long)r * VOCAB] = acc[i][j][r];
            }
        }
    }
}

// ---------------------------------------------------------------------------
// Kernel 3: fused per-row log-softmax, in place on d_out.
// One 256-thread block per row (4096 blocks). Online (max, sum-exp) per
// thread, wave32 shuffle reduce, cross-wave LDS merge, then subtract pass
// (row is 201 KB -> second read is largely L2-resident).
// ---------------------------------------------------------------------------
__global__ __launch_bounds__(256)
void row_logsoftmax_kernel(float* __restrict__ out) {
    float* p = out + (long)blockIdx.x * VOCAB;
    const int tid = threadIdx.x;

    float m = -INFINITY, s = 0.0f;
    for (int i = tid; i < VOCAB; i += 256) {
        float x = p[i];
        if (x > m) { s = s * __expf(m - x) + 1.0f; m = x; }
        else       { s += __expf(x - m); }
    }
    // wave32 reduction
#pragma unroll
    for (int off = 16; off > 0; off >>= 1) {
        float m2 = __shfl_xor(m, off, 32);
        float s2 = __shfl_xor(s, off, 32);
        float M  = fmaxf(m, m2);
        s = s * __expf(m - M) + s2 * __expf(m2 - M);
        m = M;
    }
    __shared__ float sm[8];
    __shared__ float ss[8];
    if ((tid & 31) == 0) { sm[tid >> 5] = m; ss[tid >> 5] = s; }
    __syncthreads();

    float M = sm[0], S = ss[0];
#pragma unroll
    for (int w = 1; w < 8; ++w) {
        float m2 = sm[w], s2 = ss[w];
        float Mn = fmaxf(M, m2);
        S = S * __expf(M - Mn) + s2 * __expf(m2 - Mn);
        M = Mn;
    }
    const float lse = M + logf(S);

    for (int i = tid; i < VOCAB; i += 256) p[i] -= lse;
}

// ---------------------------------------------------------------------------
// Launch. Workspace layout (bytes):
//   Whi : [VOCAB*EMBED] bf16   @ 0
//   Wlo : [VOCAB*EMBED] bf16   @ VOCAB*EMBED*2
//   Hhi : [NROWS*EMBED] bf16   @ VOCAB*EMBED*4
//   Hlo : [NROWS*EMBED] bf16   @ VOCAB*EMBED*4 + NROWS*EMBED*2
//   total ~223 MB (all offsets 16B-aligned).
// ---------------------------------------------------------------------------
extern "C" void kernel_launch(void* const* d_in, const int* in_sizes, int n_in,
                              void* d_out, int out_size, void* d_ws, size_t ws_size,
                              hipStream_t stream) {
    const float* H = (const float*)d_in[0];   // [2,2048,1024] fp32
    const float* W = (const float*)d_in[1];   // [50257,1024]  fp32
    float* out = (float*)d_out;               // [2,2048,50257] fp32

    char* ws = (char*)d_ws;
    __bf16* Whi = (__bf16*)(ws);
    __bf16* Wlo = (__bf16*)(ws + (size_t)VOCAB * EMBED * 2);
    __bf16* Hhi = (__bf16*)(ws + (size_t)VOCAB * EMBED * 4);
    __bf16* Hlo = (__bf16*)(ws + (size_t)VOCAB * EMBED * 4 + (size_t)NROWS * EMBED * 2);

    const long nW = (long)VOCAB * EMBED;
    const long nH = (long)NROWS * EMBED;
    split_bf16_kernel<<<4096, 256, 0, stream>>>(W, Whi, Wlo, nW);
    split_bf16_kernel<<<1024, 256, 0, stream>>>(H, Hhi, Hlo, nH);

    dim3 grid((VOCAB + 127) / 128, NROWS / 128);   // 393 x 32 blocks
    lmhead_gemm_kernel<<<grid, 256, 0, stream>>>(Whi, Wlo, Hhi, Hlo, out);

    row_logsoftmax_kernel<<<NROWS, 256, 0, stream>>>(out);
}
